// CausalSelfAttentionWithQKV_47708496724679
// MI455X (gfx1250) — matrix operand order repaired and checked by reading, not compiled, not run on hardware
//
#include <hip/hip_runtime.h>
#include <hip/hip_bf16.h>

// ---- problem constants (match reference) ----
constexpr int Bq   = 2;
constexpr int S    = 2048;
constexpr int H    = 1024;
constexpr int NH   = 16;
constexpr int HD   = 64;      // head dim
constexpr int N3H  = 3 * H;   // 3072
constexpr int M    = Bq * S;  // 4096 tokens

typedef __attribute__((ext_vector_type(8)))  __bf16 v8bf;
typedef __attribute__((ext_vector_type(16))) __bf16 v16bf;
typedef __attribute__((ext_vector_type(8)))  float  v8f;
typedef __attribute__((ext_vector_type(4)))  unsigned int u32x4;
typedef __attribute__((ext_vector_type(8)))  int i32x8;
typedef __attribute__((ext_vector_type(4)))  int i32x4;

// GEMM LDS staging geometry
constexpr int KC   = 128;        // K-chunk staged per TDM op
constexpr int SROW = KC + 8;     // padded LDS row (272B -> bank = 4*r mod 64, conflict-free)

// ---------------- WMMA helpers ----------------
static __device__ __forceinline__ v8f wmma_bf16(v16bf a, v16bf b, v8f c) {
  return __builtin_amdgcn_wmma_f32_16x16x32_bf16(
      false, a, false, b, (short)0, c, false, false);
}

// A fragment (16-bit A, 16x32): lane = row M; elems e<8 -> K=kk+8h+e,
// e>=8 -> K=kk+16+8h+(e-8)  ==> two contiguous 16B chunks.
static __device__ __forceinline__ v16bf load_afrag(const __bf16* __restrict__ row,
                                                   int kk, int h) {
  v8bf lo = *reinterpret_cast<const v8bf*>(row + kk + 8 * h);
  v8bf hi = *reinterpret_cast<const v8bf*>(row + kk + 16 + 8 * h);
  return __builtin_shufflevector(lo, hi, 0,1,2,3,4,5,6,7,8,9,10,11,12,13,14,15);
}

// B fragment (32x16): lane = col N; elems e<8 -> K=kk+8h+e,
// e>=8 -> K=kk+16+8h+(e-8) ==> two contiguous 16B chunks.
static __device__ __forceinline__ v16bf load_bfrag(const __bf16* __restrict__ row,
                                                   int kk, int h) {
  v8bf lo = *reinterpret_cast<const v8bf*>(row + kk + 8 * h);
  v8bf hi = *reinterpret_cast<const v8bf*>(row + kk + 16 + 8 * h);
  return __builtin_shufflevector(lo, hi, 0,1,2,3,4,5,6,7,8,9,10,11,12,13,14,15);
}

// Same B fragment from 16B-aligned (padded) LDS rows: two ds_load_b128.
static __device__ __forceinline__ v16bf load_bfrag_lds(const __bf16* row,
                                                       int kk, int h) {
  const __bf16* p = row + kk + 8 * h;
  v8bf lo = *reinterpret_cast<const v8bf*>(p);
  v8bf hi = *reinterpret_cast<const v8bf*>(p + 16);
  return __builtin_shufflevector(lo, hi, 0,1,2,3,4,5,6,7,8,9,10,11,12,13,14,15);
}

// ---------------- Tensor Data Mover staging ----------------
#if __has_builtin(__builtin_amdgcn_tensor_load_to_lds)
#define HAVE_TDM 1
#else
#define HAVE_TDM 0
#endif

#if HAVE_TDM
// 2D tile load: tile_dim0=KC elems (x), tile_dim1=64 rows, bf16 (data_size=1),
// LDS padding: 16B every 256B row (pad_interval code 5 = 64 DWORDs, amount code 3 = 4 DWORDs)
static __device__ __forceinline__ void tdm_load_64xKC(unsigned lds_off,
                                                      const __bf16* g, int pitch) {
  unsigned long long ga = (unsigned long long)(uintptr_t)g;
  u32x4 g0;
  g0[0] = 1u;                                              // count=1 (valid user D#)
  g0[1] = lds_off;                                         // lds_addr
  g0[2] = (unsigned)ga;                                    // global_addr[31:0]
  g0[3] = (unsigned)((ga >> 32) & 0x01ffffffu) | (2u << 30); // addr[56:32] | type=2
  i32x8 g1;
  g1[0] = (1 << 16) | (1 << 20) | (5 << 22) | (3 << 25);   // data_size=2B, pad enabled
  g1[1] = (pitch & 0xffff) << 16;                          // tensor_dim0[15:0]
  g1[2] = ((unsigned)pitch >> 16) & 0xffff;                // tensor_dim0[31:16]; dim1[15:0]=0
  g1[2] |= (int)((1u << 16) << 16);                        // tensor_dim1 low16 = 0x0000 -> set below
  g1[2] = (int)(((unsigned)pitch >> 16) & 0xffffu);        // dim0 hi; tensor_dim1 = 1<<20:
  g1[3] = (int)((1u << 20) >> 16) | ((KC & 0xffff) << 16); // dim1 hi16 (=16), tile_dim0=KC
  g1[4] = 64;                                              // tile_dim1=64 rows; tile_dim2=0
  g1[5] = pitch;                                           // tensor_dim0_stride[31:0]
  g1[6] = 0;
  g1[7] = 0;
  i32x4 z4 = {0, 0, 0, 0};
#if __clang_major__ >= 23
  i32x8 z8 = {0, 0, 0, 0, 0, 0, 0, 0};
  __builtin_amdgcn_tensor_load_to_lds(g0, g1, z4, z4, z8, 0);
#else
  __builtin_amdgcn_tensor_load_to_lds(g0, g1, z4, z4, 0);
#endif
}
#endif

// Stage one 64-row x KC-col bf16 chunk of Wt into padded LDS rows.
static __device__ __forceinline__ void stage_issue(bool issuer, __bf16* sdst,
                                                   const __bf16* g, int pitch) {
#if HAVE_TDM
  if (issuer) tdm_load_64xKC((unsigned)(uintptr_t)sdst, g, pitch);
#else
  for (int idx = threadIdx.x; idx < 64 * KC; idx += (int)blockDim.x) {
    int row = idx / KC, col = idx % KC;
    sdst[row * SROW + col] = g[(size_t)row * pitch + col];
  }
#endif
}

static __device__ __forceinline__ void stage_wait(bool issuer, bool more_pending) {
#if HAVE_TDM
  if (issuer) {
    if (more_pending) __builtin_amdgcn_s_wait_tensorcnt(1);
    else              __builtin_amdgcn_s_wait_tensorcnt(0);
  }
#endif
  __syncthreads();
}

// ---------------- conversion kernels ----------------
__global__ void k_cvt_f32_to_bf16(const float* __restrict__ in,
                                  __bf16* __restrict__ out, int n) {
  int i = blockIdx.x * blockDim.x + threadIdx.x;
  if (i < n) out[i] = (__bf16)in[i];
}

__global__ void k_transpose_to_bf16(const float* __restrict__ W,
                                    __bf16* __restrict__ Wt, int K, int N) {
  int i = blockIdx.x * blockDim.x + threadIdx.x;
  if (i >= K * N) return;
  int k = i / N, n = i % N;
  Wt[(size_t)n * K + k] = (__bf16)W[i];
}

// ---------------- TDM double-buffered 128x64-per-block GEMM mainloop ----------------
// Block = 8 waves; all share one 64-row B strip (Wt rows n0..n0+63) staged in LDS.
// Wave w computes rows m0..m0+15 of C into acc[0..3] (cols n0+16t .. n0+16t+15).
static __device__ __forceinline__ void gemm_mainloop(
    const __bf16* __restrict__ A, const __bf16* __restrict__ Wt, __bf16* bs,
    int m0, int n0, int K, int r, int h, v8f acc[4]) {
  const int NC = K / KC;
  const bool issuer = (threadIdx.x < 32);
  const int bufelems = 64 * SROW;

  stage_issue(issuer, bs, Wt + (size_t)n0 * K, K);                 // chunk 0
  for (int c = 0; c < NC; ++c) {
    if (c + 1 < NC)
      stage_issue(issuer, bs + ((c + 1) & 1) * bufelems,
                  Wt + (size_t)n0 * K + (c + 1) * KC, K);
    stage_wait(issuer, c + 1 < NC);                                // chunk c resident
    const __bf16* arow = A + (size_t)(m0 + r) * K + c * KC;
    const __bf16* bbuf = bs + (c & 1) * bufelems;
#pragma unroll
    for (int kk = 0; kk < KC; kk += 32) {
      v16bf av = load_afrag(arow, kk, h);
#pragma unroll
      for (int t = 0; t < 4; ++t) {
        v16bf bv = load_bfrag_lds(bbuf + (size_t)(t * 16 + r) * SROW, kk, h);
        acc[t] = wmma_bf16(av, bv, acc[t]);
      }
    }
    __syncthreads();                // compute done before buffer c&1 is refilled
  }
}

// ---------------- kernel 1: QKV projection + scatter to [B,h,S,d] ----------------
__global__ void k_gemm_qkv(const __bf16* __restrict__ Abf,     // [M,H] bf16 x
                           const __bf16* __restrict__ Wt,      // [3H,H] bf16 W_qkv^T
                           const float*  __restrict__ bias,    // [3H]
                           float* __restrict__ out_q, float* __restrict__ out_k,
                           float* __restrict__ out_v,          // fp32 [B,h,S,d]
                           __bf16* __restrict__ q_bf,          // [B,h,S,d]
                           __bf16* __restrict__ k_bf,          // [B,h,S,d]
                           __bf16* __restrict__ vt_bf) {       // [B,h,d,S]
  __shared__ __align__(64) __bf16 bs[2 * 64 * SROW];
  constexpr int NSTRIP = N3H / 64;                 // 48
  int lane = threadIdx.x & 31, r = lane & 15, h = lane >> 4;
  int wave = threadIdx.x >> 5;
  int m0 = (blockIdx.x / NSTRIP) * 128 + wave * 16;
  int n0 = (blockIdx.x % NSTRIP) * 64;

  v8f acc[4] = {v8f{}, v8f{}, v8f{}, v8f{}};
  gemm_mainloop(Abf, Wt, bs, m0, n0, H, r, h, acc);

#pragma unroll
  for (int t = 0; t < 4; ++t) {
    int n = n0 + t * 16 + r;
    int part = n / H, nn = n % H;
    int head = nn >> 6, dcol = nn & 63;
    float bv = bias[n];
#pragma unroll
    for (int e = 0; e < 8; ++e) {
      int m = m0 + e + 8 * h;
      int b = m >> 11, s = m & (S - 1);
      float val = acc[t][e] + bv;
      size_t idx = (((size_t)(b * NH + head) * S) + s) * HD + dcol;
      if (part == 0) { out_q[idx] = val; q_bf[idx] = (__bf16)val; }
      else if (part == 1) { out_k[idx] = val; k_bf[idx] = (__bf16)val; }
      else {
        out_v[idx] = val;
        vt_bf[((size_t)(b * NH + head) * HD + dcol) * S + s] = (__bf16)val; // V^T
      }
    }
  }
}

// ---------------- kernel 2: flash attention (S^T trick, all-register) ----------------
__global__ void k_attention(const __bf16* __restrict__ q_bf,   // [B,h,S,d]
                            const __bf16* __restrict__ k_bf,   // [B,h,S,d]
                            const __bf16* __restrict__ vt_bf,  // [B,h,d,S]
                            const int*    __restrict__ amask,  // [B,S]
                            __bf16* __restrict__ ctx_bf) {     // [B,S,H]
  int wid  = (blockIdx.x * blockDim.x + threadIdx.x) >> 5;
  int lane = threadIdx.x & 31, r = lane & 15, h = lane >> 4;
  constexpr int RB = S / 16;                       // 128 row blocks
  int bh = wid >> 7, iblk = wid & (RB - 1);
  if (bh >= Bq * NH) return;
  int i0 = iblk * 16;

  const __bf16* qbase = q_bf + (size_t)bh * S * HD;
  const __bf16* kbase = k_bf + (size_t)bh * S * HD;
  const __bf16* vbase = vt_bf + (size_t)bh * HD * S;
  const int* am = amask + (bh >> 4) * S;

  const __bf16* qrow = qbase + (size_t)(i0 + r) * HD;   // Q^T = B operand
  v16bf qf0 = load_bfrag(qrow, 0, h);
  v16bf qf1 = load_bfrag(qrow, 32, h);

  v8f o[4] = {v8f{}, v8f{}, v8f{}, v8f{}};   // ctx: col d = 16*dt + r, row s = e + 8h
  float m_run = -__builtin_inff(), l_run = 0.f;
  int s_glob = i0 + r;
  const float scale = 0.125f;                // 1/sqrt(64)

  int njb = (i0 + 15) / 32 + 1;              // causal, 32-wide col blocks
  for (int j = 0; j < njb; ++j) {
    int t0 = j * 32;
    v8f st0 = v8f{}, st1 = v8f{};            // S^T tiles: row t, col s
    {
      const __bf16* krow = kbase + (size_t)(t0 + r) * HD;
      st0 = wmma_bf16(load_afrag(krow, 0, h), qf0, st0);
      st0 = wmma_bf16(load_afrag(krow, 32, h), qf1, st0);
    }
    {
      const __bf16* krow = kbase + (size_t)(t0 + 16 + r) * HD;
      st1 = wmma_bf16(load_afrag(krow, 0, h), qf0, st1);
      st1 = wmma_bf16(load_afrag(krow, 32, h), qf1, st1);
    }
    float mloc = -__builtin_inff();
#pragma unroll
    for (int e = 0; e < 8; ++e) {
      int ta = t0 + e + 8 * h, tb = ta + 16;
      st0[e] = (ta <= s_glob && am[ta] != 0) ? st0[e] * scale : -__builtin_inff();
      st1[e] = (tb <= s_glob && am[tb] != 0) ? st1[e] * scale : -__builtin_inff();
      mloc = fmaxf(mloc, fmaxf(st0[e], st1[e]));
    }
    mloc = fmaxf(mloc, __shfl_xor(mloc, 16, 32));
    float m_new = fmaxf(m_run, mloc);
    float alpha = __expf(m_run - m_new);
    float ssum = 0.f;
#pragma unroll
    for (int e = 0; e < 8; ++e) {
      float p0 = __expf(st0[e] - m_new);
      float p1 = __expf(st1[e] - m_new);
      st0[e] = p0; st1[e] = p1; ssum += p0 + p1;
    }
    ssum += __shfl_xor(ssum, 16, 32);
    l_run = l_run * alpha + ssum;
    m_run = m_new;
    v16bf pa;                                 // P A-frag: lane-local repack of P^T tiles
#pragma unroll
    for (int e = 0; e < 8; ++e) { pa[e] = (__bf16)st0[e]; pa[e + 8] = (__bf16)st1[e]; }
#pragma unroll
    for (int e = 0; e < 8; ++e) {
      float ae = __shfl(alpha, e + 8 * h, 32);  // per-row rescale via bpermute
      o[0][e] *= ae; o[1][e] *= ae; o[2][e] *= ae; o[3][e] *= ae;
    }
#pragma unroll
    for (int dt = 0; dt < 4; ++dt) {
      const __bf16* vrow = vbase + (size_t)(dt * 16 + r) * S;
      o[dt] = wmma_bf16(pa, load_bfrag(vrow, t0, h), o[dt]);
    }
  }
  int b = bh >> 4, head = bh & 15;
#pragma unroll
  for (int e = 0; e < 8; ++e) {
    float le = __shfl(l_run, e + 8 * h, 32);
    float inv = 1.0f / le;
    int s = i0 + e + 8 * h;
    size_t tok = (size_t)b * S + s;
#pragma unroll
    for (int dt = 0; dt < 4; ++dt) {
      size_t col = (size_t)head * HD + dt * 16 + r;
      ctx_bf[tok * H + col] = (__bf16)(o[dt][e] * inv);
    }
  }
}

// ---------------- kernel 3: output projection ----------------
__global__ void k_gemm_out(const __bf16* __restrict__ ctx_bf,  // [M,H]
                           const __bf16* __restrict__ Wt,      // [H,H] W_out^T
                           const float*  __restrict__ bias,    // [H]
                           float* __restrict__ out) {          // [M,H]
  __shared__ __align__(64) __bf16 bs[2 * 64 * SROW];
  constexpr int NSTRIP = H / 64;                   // 16
  int lane = threadIdx.x & 31, r = lane & 15, h = lane >> 4;
  int wave = threadIdx.x >> 5;
  int m0 = (blockIdx.x / NSTRIP) * 128 + wave * 16;
  int n0 = (blockIdx.x % NSTRIP) * 64;

  v8f acc[4] = {v8f{}, v8f{}, v8f{}, v8f{}};
  gemm_mainloop(ctx_bf, Wt, bs, m0, n0, H, r, h, acc);

#pragma unroll
  for (int t = 0; t < 4; ++t) {
    int n = n0 + t * 16 + r;
    float bv = bias[n];
#pragma unroll
    for (int e = 0; e < 8; ++e) {
      int m = m0 + e + 8 * h;
      out[(size_t)m * H + n] = acc[t][e] + bv;
    }
  }
}

// ---------------- launcher ----------------
extern "C" void kernel_launch(void* const* d_in, const int* in_sizes, int n_in,
                              void* d_out, int out_size, void* d_ws, size_t ws_size,
                              hipStream_t stream) {
  (void)in_sizes; (void)n_in; (void)out_size; (void)ws_size;
  const float* x     = (const float*)d_in[0];
  const int*   amask = (const int*)d_in[1];
  const float* Wqkv  = (const float*)d_in[2];
  const float* bqkv  = (const float*)d_in[3];
  const float* Wout  = (const float*)d_in[4];
  const float* bout  = (const float*)d_in[5];

  float* out   = (float*)d_out;                       // [B,S,H]
  float* out_q = out   + (size_t)M * H;               // [B,h,S,d]
  float* out_k = out_q + (size_t)M * H;
  float* out_v = out_k + (size_t)M * H;

  // workspace layout (~48 MB)
  char* p = (char*)d_ws;
  __bf16* x_bf   = (__bf16*)p; p += (size_t)M * H * 2;
  __bf16* wqkv_t = (__bf16*)p; p += (size_t)N3H * H * 2;
  __bf16* wout_t = (__bf16*)p; p += (size_t)H * H * 2;
  __bf16* q_bf   = (__bf16*)p; p += (size_t)M * H * 2;
  __bf16* k_bf   = (__bf16*)p; p += (size_t)M * H * 2;
  __bf16* vt_bf  = (__bf16*)p; p += (size_t)M * H * 2;
  __bf16* ctx_bf = (__bf16*)p; p += (size_t)M * H * 2;

  k_cvt_f32_to_bf16<<<(M * H) / 256, 256, 0, stream>>>(x, x_bf, M * H);
  k_transpose_to_bf16<<<(H * N3H) / 256, 256, 0, stream>>>(Wqkv, wqkv_t, H, N3H);
  k_transpose_to_bf16<<<(H * H) / 256, 256, 0, stream>>>(Wout, wout_t, H, H);

  // 1) QKV projection: 128x64 C-tile per block (8 waves, shared TDM-staged B)
  k_gemm_qkv<<<(M / 128) * (N3H / 64), 256, 0, stream>>>(
      x_bf, wqkv_t, bqkv, out_q, out_k, out_v, q_bf, k_bf, vt_bf);

  // 2) flash attention: one wave per (b, head, 16-row block) = 4096 waves
  k_attention<<<(Bq * NH * (S / 16)) / 8, 256, 0, stream>>>(
      q_bf, k_bf, vt_bf, amask, ctx_bf);

  // 3) output projection
  k_gemm_out<<<(M / 128) * (H / 64), 256, 0, stream>>>(
      ctx_bf, wout_t, bout, out);
}
